// BKT_RNN_84877143703787
// MI455X (gfx1250) — compile-verified, matrix operand order checked by reading, not run on hardware
//
#include <hip/hip_runtime.h>
#include <math.h>

// ---------------------------------------------------------------------------
// Problem constants (from reference)
// ---------------------------------------------------------------------------
#define S_   256
#define B_   64
#define XSZ  4367
#define D_   512
#define H_   8
#define DH_  64
#define FF_  2048
#define L_   6
#define HID_ 4
#define M_ROWS (S_ * B_)      // 16384

#define BPAD 40               // LDS row stride in bf16 (32 data + 8 pad, 80B)

typedef __attribute__((ext_vector_type(16))) __bf16 v16bf;
typedef __attribute__((ext_vector_type(8)))  float  v8f;

// ---------------------------------------------------------------------------
// Helpers
// ---------------------------------------------------------------------------
__device__ __forceinline__ float wave_sum(float v) {
#pragma unroll
  for (int m = 16; m >= 1; m >>= 1) v += __shfl_xor(v, m, 32);
  return v;
}
__device__ __forceinline__ float wave_max(float v) {
#pragma unroll
  for (int m = 16; m >= 1; m >>= 1) v = fmaxf(v, __shfl_xor(v, m, 32));
  return v;
}

// unguarded: 8 consecutive f32 -> bf16
__device__ __forceinline__ void cvt8_fast(const float* __restrict__ p, int k,
                                          __bf16* dst) {
#pragma unroll
  for (int i = 0; i < 8; ++i) dst[i] = (__bf16)p[k + i];
}
// guarded (zero-pad past K) — used only in the single tail step
__device__ __forceinline__ void cvt8_guard(const float* __restrict__ p, int k, int K,
                                           __bf16* dst) {
#pragma unroll
  for (int i = 0; i < 8; ++i) dst[i] = (k + i < K) ? (__bf16)p[k + i] : (__bf16)0.0f;
}

// ---------------------------------------------------------------------------
// f32 -> bf16 weight conversion (grid-stride)
// ---------------------------------------------------------------------------
__global__ void cvt_f32_bf16(const float* __restrict__ s, __bf16* __restrict__ d, int n) {
  int i = blockIdx.x * blockDim.x + threadIdx.x;
  int stride = gridDim.x * blockDim.x;
  for (; i < n; i += stride) d[i] = (__bf16)s[i];
}

// ---------------------------------------------------------------------------
// Cooperative fill of one 64x32 bf16 B tile into LDS.
// 128 threads, thread t loads 16 bf16 (32B): col = t>>1, k-half = (t&1)*16.
// Guarded against K tail (cheap: runs once per K-step, not per fragment).
// ---------------------------------------------------------------------------
__device__ __forceinline__ void fill_b(__bf16 (*__restrict__ dst)[BPAD],
                                       const __bf16* __restrict__ Bw,
                                       int n0, int K, int kb, int tid) {
  const int col = tid >> 1;
  const int kh  = (tid & 1) * 16;
  const __bf16* src = Bw + (size_t)(n0 + col) * K + kb + kh;
  __bf16* d = &dst[col][kh];
  const int kg = kb + kh;
  if (kg + 16 <= K) {
#pragma unroll
    for (int i = 0; i < 16; ++i) d[i] = src[i];
  } else {
#pragma unroll
    for (int i = 0; i < 16; ++i) d[i] = (kg + i < K) ? src[i] : (__bf16)0.0f;
  }
}

// One K-step of the wave-tile GEMM: A from global (f32->bf16), B from LDS.
__device__ __forceinline__ void gemm_step(const float* __restrict__ A, int K,
                                          int m0, int lidx, int khalf, int kb,
                                          bool tail,
                                          const __bf16 (*__restrict__ bsh)[BPAD],
                                          v8f acc[4][4]) {
  v16bf af[4];
#pragma unroll
  for (int mt = 0; mt < 4; ++mt) {
    const float* ar = A + (size_t)(m0 + mt * 16 + lidx) * K;
    __bf16 t[16];
    if (!tail) {
      cvt8_fast(ar, kb + khalf * 8, t);
      cvt8_fast(ar, kb + 16 + khalf * 8, t + 8);
    } else {
      cvt8_guard(ar, kb + khalf * 8, K, t);
      cvt8_guard(ar, kb + 16 + khalf * 8, K, t + 8);
    }
#pragma unroll
    for (int i = 0; i < 16; ++i) af[mt][i] = t[i];
  }
  v16bf bf[4];
#pragma unroll
  for (int nt = 0; nt < 4; ++nt) {
    const __bf16* b0 = &bsh[nt * 16 + lidx][khalf * 8];
    const __bf16* b1 = &bsh[nt * 16 + lidx][16 + khalf * 8];
    __bf16 t[16];
#pragma unroll
    for (int i = 0; i < 8; ++i) { t[i] = b0[i]; t[8 + i] = b1[i]; }
#pragma unroll
    for (int i = 0; i < 16; ++i) bf[nt][i] = t[i];
  }
#pragma unroll
  for (int mt = 0; mt < 4; ++mt)
#pragma unroll
    for (int nt = 0; nt < 4; ++nt)
      acc[mt][nt] = __builtin_amdgcn_wmma_f32_16x16x32_bf16(
          false, af[mt], false, bf[nt], (short)0, acc[mt][nt], false, false);
}

// ---------------------------------------------------------------------------
// Generic GEMM:  C[M,N] = act(A[M,K](f32) * Bw[N,K]^T(bf16) + bias[N])
// Block = 4 waves; wave tile 64x64 (4x4 WMMA accs); B tile double-buffered
// through LDS (shared by all 4 waves -> 4x less global B traffic).
// act: 0 = none, 1 = relu
// ---------------------------------------------------------------------------
__global__ __launch_bounds__(128) void gemm_bf16_wmma(
    const float* __restrict__ A, const __bf16* __restrict__ Bw,
    const float* __restrict__ bias, float* __restrict__ C,
    int M, int N, int K, int act) {
  __shared__ __align__(16) __bf16 bsh[2][64][BPAD];
  const int tid   = threadIdx.x;
  const int wave  = tid >> 5;
  const int lane  = tid & 31;
  const int lidx  = lane & 15;
  const int khalf = lane >> 4;
  const int m0 = (blockIdx.y * 4 + wave) * 64;
  const int n0 = blockIdx.x * 64;

  v8f acc[4][4];
#pragma unroll
  for (int i = 0; i < 4; ++i)
#pragma unroll
    for (int j = 0; j < 4; ++j)
#pragma unroll
      for (int r = 0; r < 8; ++r) acc[i][j][r] = 0.0f;

  fill_b(bsh[0], Bw, n0, K, 0, tid);
  __syncthreads();

  int buf = 0;
  int kb = 0;
  for (; kb + 32 <= K; kb += 32) {          // guard-free main loop
    const int nxt = buf ^ 1;
    if (kb + 32 < K) fill_b(bsh[nxt], Bw, n0, K, kb + 32, tid);
    gemm_step(A, K, m0, lidx, khalf, kb, /*tail=*/false, bsh[buf], acc);
    __syncthreads();
    buf = nxt;
  }
  if (kb < K) {                             // single guarded tail step
    gemm_step(A, K, m0, lidx, khalf, kb, /*tail=*/true, bsh[buf], acc);
  }

#pragma unroll
  for (int nt = 0; nt < 4; ++nt) {
    const int n = n0 + nt * 16 + lidx;
    const float bv = bias[n];
#pragma unroll
    for (int mt = 0; mt < 4; ++mt) {
      const int mrow = m0 + mt * 16 + khalf * 8;
#pragma unroll
      for (int r = 0; r < 8; ++r) {
        float v = acc[mt][nt][r] + bv;
        if (act == 1) v = fmaxf(v, 0.0f);
        C[(size_t)(mrow + r) * N + n] = v;
      }
    }
  }
}

// ---------------------------------------------------------------------------
// Attention scores: per (b,h) compute Q(256x64) * K(256x64)^T / 8 via WMMA.
// Rows of Q/K live in the qkv buffer (S,B,3D) with row stride B*3D.
// grid = (4, 1, B*H); block = 128 (4 waves cover M=256). K=64: no guards.
// ---------------------------------------------------------------------------
__global__ __launch_bounds__(128) void attn_scores_wmma(
    const float* __restrict__ qkv, float* __restrict__ sc) {
  const int wave  = threadIdx.x >> 5;
  const int lane  = threadIdx.x & 31;
  const int lidx  = lane & 15;
  const int khalf = lane >> 4;
  const int z  = blockIdx.z;      // b*H + h
  const int b  = z >> 3;
  const int hh = z & 7;
  const int m0 = wave * 64;
  const int n0 = blockIdx.x * 64;
  const size_t ld = (size_t)B_ * 3 * D_;  // 98304
  const float* qbase = qkv + (size_t)b * (3 * D_) + hh * DH_;
  const float* kbase = qbase + D_;

  v8f acc[4][4];
#pragma unroll
  for (int i = 0; i < 4; ++i)
#pragma unroll
    for (int j = 0; j < 4; ++j)
#pragma unroll
      for (int r = 0; r < 8; ++r) acc[i][j][r] = 0.0f;

#pragma unroll
  for (int kb = 0; kb < DH_; kb += 32) {
    v16bf af[4], bf[4];
#pragma unroll
    for (int mt = 0; mt < 4; ++mt) {
      const float* ar = qbase + (size_t)(m0 + mt * 16 + lidx) * ld;
      __bf16 t[16];
      cvt8_fast(ar, kb + khalf * 8, t);
      cvt8_fast(ar, kb + 16 + khalf * 8, t + 8);
#pragma unroll
      for (int i = 0; i < 16; ++i) af[mt][i] = t[i];
    }
#pragma unroll
    for (int nt = 0; nt < 4; ++nt) {
      const float* br = kbase + (size_t)(n0 + nt * 16 + lidx) * ld;
      __bf16 t[16];
      cvt8_fast(br, kb + khalf * 8, t);
      cvt8_fast(br, kb + 16 + khalf * 8, t + 8);
#pragma unroll
      for (int i = 0; i < 16; ++i) bf[nt][i] = t[i];
    }
#pragma unroll
    for (int mt = 0; mt < 4; ++mt)
#pragma unroll
      for (int nt = 0; nt < 4; ++nt)
        acc[mt][nt] = __builtin_amdgcn_wmma_f32_16x16x32_bf16(
            false, af[mt], false, bf[nt], (short)0, acc[mt][nt], false, false);
  }

#pragma unroll
  for (int nt = 0; nt < 4; ++nt) {
    const int n = n0 + nt * 16 + lidx;
#pragma unroll
    for (int mt = 0; mt < 4; ++mt) {
      const int mrow = m0 + mt * 16 + khalf * 8;
#pragma unroll
      for (int r = 0; r < 8; ++r)
        sc[((size_t)z * S_ + (mrow + r)) * S_ + n] = acc[mt][nt][r] * 0.125f;
    }
  }
}

// ---------------------------------------------------------------------------
// Softmax over last dim (256) of scores; wave per row; in place.
// ---------------------------------------------------------------------------
__global__ __launch_bounds__(256) void softmax256(float* __restrict__ sc) {
  const int wave = threadIdx.x >> 5;
  const int lane = threadIdx.x & 31;
  float* p = sc + ((size_t)blockIdx.x * 8 + wave) * S_;
  float v[8];
  float mx = -3.0e38f;
#pragma unroll
  for (int j = 0; j < 8; ++j) { v[j] = p[lane + 32 * j]; mx = fmaxf(mx, v[j]); }
  mx = wave_max(mx);
  float s = 0.0f;
#pragma unroll
  for (int j = 0; j < 8; ++j) { v[j] = expf(v[j] - mx); s += v[j]; }
  s = wave_sum(s);
  const float inv = 1.0f / s;
#pragma unroll
  for (int j = 0; j < 8; ++j) p[lane + 32 * j] = v[j] * inv;
}

// ---------------------------------------------------------------------------
// attn @ V -> o laid out as (S,B,D) f32 (ready to be the next GEMM's A).
// grid = S*B*H blocks; 64 threads = dh.  (~0.6% of total FLOPs, VALU is fine)
// ---------------------------------------------------------------------------
__global__ __launch_bounds__(64) void attn_av(
    const float* __restrict__ attn, const float* __restrict__ qkv,
    float* __restrict__ o) {
  const int s  = blockIdx.x >> 9;
  const int b  = (blockIdx.x >> 3) & 63;
  const int hh = blockIdx.x & 7;
  const int dh = threadIdx.x;
  const float* ar = attn + ((size_t)(b * H_ + hh) * S_ + s) * S_;
  const float* vb = qkv + (size_t)b * (3 * D_) + 2 * D_ + hh * DH_ + dh;
  float acc = 0.0f;
#pragma unroll 4
  for (int k = 0; k < S_; ++k)
    acc = fmaf(ar[k], vb[(size_t)k * (B_ * 3 * D_)], acc);
  o[((size_t)(s * B_ + b)) * D_ + hh * DH_ + dh] = acc;
}

// ---------------------------------------------------------------------------
// out = LayerNorm(A + Bv) * g + beta ; wave per row of 512; safe in place.
// ---------------------------------------------------------------------------
__global__ __launch_bounds__(256) void residual_ln(
    const float* __restrict__ A, const float* __restrict__ Bv,
    const float* __restrict__ g, const float* __restrict__ be,
    float* __restrict__ out) {
  const int wave = threadIdx.x >> 5;
  const int lane = threadIdx.x & 31;
  const size_t row = (size_t)blockIdx.x * 8 + wave;
  const float* a = A + row * D_;
  const float* bb = Bv + row * D_;
  float x[16], s1 = 0.0f, s2 = 0.0f;
#pragma unroll
  for (int j = 0; j < 16; ++j) {
    const int i = lane + 32 * j;
    const float v = a[i] + bb[i];
    x[j] = v; s1 += v; s2 += v * v;
  }
  s1 = wave_sum(s1); s2 = wave_sum(s2);
  const float mu  = s1 * (1.0f / D_);
  const float var = s2 * (1.0f / D_) - mu * mu;
  const float r   = rsqrtf(var + 1e-5f);
  float* o = out + row * D_;
#pragma unroll
  for (int j = 0; j < 16; ++j) {
    const int i = lane + 32 * j;
    o[i] = (x[j] - mu) * r * g[i] + be[i];
  }
}

// ---------------------------------------------------------------------------
// params[row, j] = sigmoid(h[row,:] . post_w[j,:] + post_b[j]); wave per output.
// ---------------------------------------------------------------------------
__global__ __launch_bounds__(256) void post_head(
    const float* __restrict__ h, const float* __restrict__ pw,
    const float* __restrict__ pb, float* __restrict__ params) {
  const int wave = threadIdx.x >> 5;
  const int lane = threadIdx.x & 31;
  const int g = blockIdx.x * 8 + wave;       // 0 .. 65535
  const int row = g >> 2, j = g & 3;
  const float* hr = h + (size_t)row * D_;
  const float* wr = pw + j * D_;
  float s = 0.0f;
#pragma unroll
  for (int t = 0; t < 16; ++t) {
    const int i = lane + 32 * t;
    s = fmaf(hr[i], wr[i], s);
  }
  s = wave_sum(s);
  if (lane == 0) params[row * HID_ + j] = 1.0f / (1.0f + expf(-(s + pb[j])));
}

// ---------------------------------------------------------------------------
// BKT sequential scan over S; thread per batch element; loss reduction.
// out layout: [0,16384) corrects, [16384,32768) latents, [32768] loss.
// ---------------------------------------------------------------------------
__global__ __launch_bounds__(64) void bkt_scan(
    const float* __restrict__ params, const float* __restrict__ y,
    const float* __restrict__ prior, float* __restrict__ out) {
  const int b = threadIdx.x;
  float latent = 1.0f / (1.0f + expf(-prior[0]));
  float loss = 0.0f;
  for (int s = 0; s < S_; ++s) {
    const float* p = params + ((size_t)s * B_ + b) * HID_;
    const float l = p[0], f = p[1], g = p[2], sl = p[3];
    const float c  = latent * (1.0f - sl) + (1.0f - latent) * g;
    const float k1 = latent * (1.0f - sl) / c;
    const float k0 = latent * sl / (latent * sl + (1.0f - latent) * (1.0f - g));
    const float m  = k1 * c + k0 * (1.0f - c);
    const float nl = m * (1.0f - f) + (1.0f - m) * l;
    out[s * B_ + b] = c;
    out[S_ * B_ + s * B_ + b] = nl;
    const float yy = y[s * B_ + b];
    loss -= yy * logf(c) + (1.0f - yy) * logf(1.0f - c);
    latent = nl;
  }
  loss = wave_sum(loss);
  __shared__ float part[2];
  if ((b & 31) == 0) part[b >> 5] = loss;
  __syncthreads();
  if (b == 0) out[2 * S_ * B_] = (part[0] + part[1]) * (1.0f / (S_ * B_));
}

// ---------------------------------------------------------------------------
// Host orchestration
// ---------------------------------------------------------------------------
extern "C" void kernel_launch(void* const* d_in, const int* in_sizes, int n_in,
                              void* d_out, int out_size, void* d_ws, size_t ws_size,
                              hipStream_t stream) {
  (void)in_sizes; (void)n_in; (void)out_size; (void)ws_size;
  const float* x      = (const float*)d_in[0];
  const float* y      = (const float*)d_in[1];
  const float* prior  = (const float*)d_in[2];
  const float* pre_w  = (const float*)d_in[3];
  const float* pre_b  = (const float*)d_in[4];
  const float* qkv_w  = (const float*)d_in[5];
  const float* qkv_b  = (const float*)d_in[6];
  const float* out_w  = (const float*)d_in[7];
  const float* out_b  = (const float*)d_in[8];
  const float* ff1_w  = (const float*)d_in[9];
  const float* ff1_b  = (const float*)d_in[10];
  const float* ff2_w  = (const float*)d_in[11];
  const float* ff2_b  = (const float*)d_in[12];
  const float* ln1_g  = (const float*)d_in[13];
  const float* ln1_b  = (const float*)d_in[14];
  const float* ln2_g  = (const float*)d_in[15];
  const float* ln2_b  = (const float*)d_in[16];
  const float* post_w = (const float*)d_in[17];
  const float* post_b = (const float*)d_in[18];

  // workspace carve-up (floats)
  float* wsf  = (float*)d_ws;
  float* h    = wsf;                                  //  16384*512
  float* qkvb = h    + (size_t)M_ROWS * D_;           //  16384*1536
  float* big  = qkvb + (size_t)M_ROWS * 3 * D_;       //  33,554,432 f (scores / proj / ff)
  float* obuf = big  + (size_t)33554432;              //  16384*512
  float* par  = obuf + (size_t)M_ROWS * D_;           //  16384*4
  __bf16* wpre = (__bf16*)(par + (size_t)M_ROWS * HID_);
  __bf16* wqkv = wpre + (size_t)D_ * XSZ;             // 2,235,904
  __bf16* wout = wqkv + (size_t)L_ * 3 * D_ * D_;     // 4,718,592
  __bf16* wff1 = wout + (size_t)L_ * D_ * D_;         // 1,572,864
  __bf16* wff2 = wff1 + (size_t)L_ * FF_ * D_;        // 6,291,456

  // weights -> bf16 (idempotent, graph-safe)
  cvt_f32_bf16<<<4096, 256, 0, stream>>>(pre_w, wpre, D_ * XSZ);
  cvt_f32_bf16<<<4096, 256, 0, stream>>>(qkv_w, wqkv, L_ * 3 * D_ * D_);
  cvt_f32_bf16<<<4096, 256, 0, stream>>>(out_w, wout, L_ * D_ * D_);
  cvt_f32_bf16<<<4096, 256, 0, stream>>>(ff1_w, wff1, L_ * FF_ * D_);
  cvt_f32_bf16<<<4096, 256, 0, stream>>>(ff2_w, wff2, L_ * FF_ * D_);

  // pre-projection + relu : h = relu(x @ pre_w^T + pre_b)
  gemm_bf16_wmma<<<dim3(D_ / 64, M_ROWS / 256), 128, 0, stream>>>(
      x, wpre, pre_b, h, M_ROWS, D_, XSZ, 1);

  for (int l = 0; l < L_; ++l) {
    // qkv projection
    gemm_bf16_wmma<<<dim3(3 * D_ / 64, M_ROWS / 256), 128, 0, stream>>>(
        h, wqkv + (size_t)l * 3 * D_ * D_, qkv_b + l * 3 * D_, qkvb,
        M_ROWS, 3 * D_, D_, 0);
    // scores = Q K^T / sqrt(dh)
    attn_scores_wmma<<<dim3(S_ / 64, 1, B_ * H_), 128, 0, stream>>>(qkvb, big);
    // softmax
    softmax256<<<(B_ * H_ * S_) / 8, 256, 0, stream>>>(big);
    // attn @ V
    attn_av<<<S_ * B_ * H_, 64, 0, stream>>>(big, qkvb, obuf);
    // output projection (writes proj into big; attn no longer needed)
    gemm_bf16_wmma<<<dim3(D_ / 64, M_ROWS / 256), 128, 0, stream>>>(
        obuf, wout + (size_t)l * D_ * D_, out_b + l * D_, big,
        M_ROWS, D_, D_, 0);
    // h = LN(h + proj)
    residual_ln<<<M_ROWS / 8, 256, 0, stream>>>(
        h, big, ln1_g + l * D_, ln1_b + l * D_, h);
    // ff1 + relu
    gemm_bf16_wmma<<<dim3(FF_ / 64, M_ROWS / 256), 128, 0, stream>>>(
        h, wff1 + (size_t)l * FF_ * D_, ff1_b + l * FF_, big,
        M_ROWS, FF_, D_, 1);
    // ff2
    gemm_bf16_wmma<<<dim3(D_ / 64, M_ROWS / 256), 128, 0, stream>>>(
        big, wff2 + (size_t)l * D_ * FF_, ff2_b + l * D_, obuf,
        M_ROWS, D_, FF_, 0);
    // h = LN(h + ff)
    residual_ln<<<M_ROWS / 8, 256, 0, stream>>>(
        h, obuf, ln2_g + l * D_, ln2_b + l * D_, h);
  }

  // sigmoid head -> params (S,B,4)
  post_head<<<(M_ROWS * HID_) / 8, 256, 0, stream>>>(h, post_w, post_b, par);

  // sequential BKT scan + loss
  bkt_scan<<<1, 64, 0, stream>>>(par, y, prior, (float*)d_out);
}
